// KMeansQuantizer_52553219834013
// MI455X (gfx1250) — compile-verified, matrix operand order checked by reading, not run on hardware
//
#include <hip/hip_runtime.h>
#include <stdint.h>

// ---------------------------------------------------------------------------
// KMeans nearest-centroid ids on MI455X (gfx1250, wave32, WMMA).
//
// argmin_k ( -2 x.c_k + ||c_k||^2 ),  x:[32000,1024] f32, c:[2048,1024] f32.
// 134 GFLOP GEMM vs 140 MB HBM traffic -> compute bound -> bf16 WMMA with
// hi/lo split emulation (3x v_wmma_f32_16x16x32_bf16 per tile, ~fp32 acc).
// B (centroid) tiles are staged once per block into LDS via the CDNA5 async
// copy path (ASYNCcnt) and shared by all 8 waves: cuts L2/L0 B traffic 8x
// (16 GB -> 2 GB over the kernel), keeping the matrix pipes fed.
// ---------------------------------------------------------------------------

typedef __attribute__((ext_vector_type(8)))  __bf16 v8bf;
typedef __attribute__((ext_vector_type(16))) __bf16 v16bf;
typedef __attribute__((ext_vector_type(8)))  float  v8f;
typedef __attribute__((ext_vector_type(4)))  float  v4f;

#define NROW  32000
#define EDIM  1024
#define KCENT 2048

#define NT      64              // centroid columns per n-tile
#define EC      64              // E elements staged per LDS chunk
#define BPAD    8               // LDS padding: stride 72 elems = 36 banks
#define BSTRIDE (EC + BPAD)

// workspace layout (bytes)
#define XHI_OFS   0ull
#define XLO_OFS   65536000ull                       // 32000*1024*2
#define CHI_OFS   131072000ull                      // + 32000*1024*2
#define CLO_OFS   135266304ull                      // + 2048*1024*2
#define CNORM_OFS 139460608ull                      // + 2048*1024*2
// total: 139,468,800 bytes (~139.5 MB) of d_ws

#if __has_builtin(__builtin_amdgcn_global_load_async_to_lds_b128)
#define KQ_ASYNC_LDS 1
#else
#define KQ_ASYNC_LDS 0
#endif

// Builtin parameter pointee type is a gcc-style <4 x i32> vector
// (clang diagnostic: "__attribute__((__vector_size__(4 * sizeof(int)))) int *").
typedef int kq_v4i __attribute__((vector_size(16)));
typedef __attribute__((address_space(1))) kq_v4i* kq_gvec;  // global (AS1)
typedef __attribute__((address_space(3))) kq_v4i* kq_lvec;  // LDS (AS3)

// 16-byte global -> LDS copy. Async (ASYNCcnt) when the toolchain exposes the
// gfx1250 builtin; otherwise a synchronous load+ds_store with the same net
// data movement.
__device__ __forceinline__ void kq_copy16(const __bf16* g, __bf16* l) {
#if KQ_ASYNC_LDS
    __builtin_amdgcn_global_load_async_to_lds_b128(
        (kq_gvec)(uintptr_t)g,
        (kq_lvec)(uint32_t)(uintptr_t)l,
        /*imm offset=*/0, /*cpol=*/0);
#else
    *(v8bf*)l = *(const v8bf*)g;
#endif
}

__device__ __forceinline__ void kq_stage_wait() {
#if KQ_ASYNC_LDS
#if __has_builtin(__builtin_amdgcn_s_wait_asynccnt)
    __builtin_amdgcn_s_wait_asynccnt(0);
#else
    asm volatile("s_wait_asynccnt 0x0" ::: "memory");
#endif
#endif
    __syncthreads();
}

// ---------------------------------------------------------------------------
// Prep 1: split x (f32) into bf16 hi + lo residual.
// ---------------------------------------------------------------------------
__global__ void kq_split_x(const float* __restrict__ x,
                           __bf16* __restrict__ xhi,
                           __bf16* __restrict__ xlo) {
    size_t i = ((size_t)blockIdx.x * blockDim.x + threadIdx.x) * 4;
    if (i >= (size_t)NROW * EDIM) return;
    v4f v = *(const v4f*)(x + i);
#pragma unroll
    for (int j = 0; j < 4; ++j) {
        float  f = v[j];
        __bf16 h = (__bf16)f;
        xhi[i + j] = h;
        xlo[i + j] = (__bf16)(f - (float)h);
    }
}

// ---------------------------------------------------------------------------
// Prep 2: split centroids into bf16 hi/lo and compute ||c_k||^2 (fp32).
// One wave32 per centroid row; 8 rows per 256-thread block.
// ---------------------------------------------------------------------------
__global__ void kq_split_c(const float* __restrict__ c,
                           __bf16* __restrict__ chi,
                           __bf16* __restrict__ clo,
                           float*  __restrict__ cnorm) {
    const int wave = threadIdx.x >> 5;
    const int lane = threadIdx.x & 31;
    const int row  = blockIdx.x * 8 + wave;
    if (row >= KCENT) return;
    const float* src = c + (size_t)row * EDIM;
    float ssq = 0.0f;
#pragma unroll 4
    for (int e = lane; e < EDIM; e += 32) {
        float  f = src[e];
        __bf16 h = (__bf16)f;
        chi[(size_t)row * EDIM + e] = h;
        clo[(size_t)row * EDIM + e] = (__bf16)(f - (float)h);
        ssq = fmaf(f, f, ssq);
    }
#pragma unroll
    for (int m = 16; m >= 1; m >>= 1) ssq += __shfl_xor(ssq, m, 32);
    if (lane == 0) cnorm[row] = ssq;
}

// ---------------------------------------------------------------------------
// Fragment loader: 16-bit A/B operand layout (ISA 7.12.2).
// lane&15 selects the matrix row (A) / column (B); lane>>4 selects which
// K-half this lane holds: K {0..7,16..23} (half 0) or {8..15,24..31} (half 1).
// Works for global (x rows) and LDS (staged centroid cols) alike.
// ---------------------------------------------------------------------------
__device__ __forceinline__ v16bf kq_frag(const __bf16* __restrict__ p,
                                         int o0, int o1) {
    v8bf a = *(const v8bf*)(p + o0);
    v8bf b = *(const v8bf*)(p + o1);
    return __builtin_shufflevector(a, b, 0, 1, 2, 3, 4, 5, 6, 7,
                                   8, 9, 10, 11, 12, 13, 14, 15);
}

__device__ __forceinline__ v8f kq_wmma(v16bf a, v16bf b, v8f c) {
    return __builtin_amdgcn_wmma_f32_16x16x32_bf16(
        /*neg_a=*/false, a, /*neg_b=*/false, b,
        /*c_mod=*/(short)0, c, /*reuse_a=*/false, /*reuse_b=*/false);
}

// ---------------------------------------------------------------------------
// Main: 8 waves/block, each wave owns 16 rows; block tile = 128 rows.
// N-tile = 64 centroids (4 subtiles/wave); E staged to LDS in 64-wide chunks,
// double-buffered async copies overlapping 24 WMMAs per chunk.
// ---------------------------------------------------------------------------
__global__ __launch_bounds__(256)
void kq_argmin(const __bf16* __restrict__ xhi, const __bf16* __restrict__ xlo,
               const __bf16* __restrict__ chi, const __bf16* __restrict__ clo,
               const float*  __restrict__ cnorm,
               long long* __restrict__ out) {
    __shared__ __align__(16) __bf16 sB[2][2][NT][BSTRIDE]; // [buf][hi/lo][col][e]

    const int tid     = threadIdx.x;
    const int wave    = tid >> 5;
    const int lane    = tid & 31;
    const int half    = lane >> 4;
    const int l16     = lane & 15;
    const int rowBase = blockIdx.x * 128 + wave * 16;

    const __bf16* arow_h = xhi + (size_t)(rowBase + l16) * EDIM;
    const __bf16* arow_l = xlo + (size_t)(rowBase + l16) * EDIM;

    // staging slots for this thread: 512 16B vectors per matrix per chunk,
    // 256 threads -> 2 vectors each (col = v>>3, elem = (v&7)*8)
    const int sv0 = tid * 2, sv1 = tid * 2 + 1;

    float minv[8];
    int   mini[8];
#pragma unroll
    for (int r = 0; r < 8; ++r) { minv[r] = 3.4e38f; mini[r] = 0; }

    for (int nt = 0; nt < KCENT; nt += NT) {
        v8f acc[4] = {};

        // prologue: stage chunk e=0 into buffer 0
        {
#pragma unroll
            for (int t = 0; t < 2; ++t) {
                const int v   = (t == 0) ? sv0 : sv1;
                const int col = v >> 3;
                const int ve  = (v & 7) * 8;
                const size_t g = (size_t)(nt + col) * EDIM + ve;
                kq_copy16(chi + g, &sB[0][0][col][ve]);
                kq_copy16(clo + g, &sB[0][1][col][ve]);
            }
        }
        kq_stage_wait();

        for (int ec = 0; ec < EDIM; ec += EC) {
            const int cur = (ec / EC) & 1;
            // stage next chunk into the other buffer (free since the barrier
            // that ended the previous iteration)
            if (ec + EC < EDIM) {
#pragma unroll
                for (int t = 0; t < 2; ++t) {
                    const int v   = (t == 0) ? sv0 : sv1;
                    const int col = v >> 3;
                    const int ve  = (v & 7) * 8;
                    const size_t g = (size_t)(nt + col) * EDIM + (ec + EC) + ve;
                    kq_copy16(chi + g, &sB[cur ^ 1][0][col][ve]);
                    kq_copy16(clo + g, &sB[cur ^ 1][1][col][ve]);
                }
            }

            // compute on current buffer: two 32-wide K slices
#pragma unroll
            for (int sub = 0; sub < EC; sub += 32) {
                const int e  = ec + sub;
                const int o0 = sub + half * 8;
                const int o1 = sub + 16 + half * 8;
                __builtin_prefetch(arow_h + e + 128, 0, 1);

                v16bf a_h = kq_frag(arow_h, e + half * 8, e + 16 + half * 8);
                v16bf a_l = kq_frag(arow_l, e + half * 8, e + 16 + half * 8);
#pragma unroll
                for (int s = 0; s < 4; ++s) {
                    const __bf16* bh = &sB[cur][0][s * 16 + l16][0];
                    const __bf16* bl = &sB[cur][1][s * 16 + l16][0];
                    v16bf b_h = kq_frag(bh, o0, o1);
                    v16bf b_l = kq_frag(bl, o0, o1);
                    acc[s] = kq_wmma(a_h, b_h, acc[s]);
                    acc[s] = kq_wmma(a_h, b_l, acc[s]);
                    acc[s] = kq_wmma(a_l, b_h, acc[s]);
                }
            }
            kq_stage_wait();   // next buffer complete; all waves done reading
        }

        // fold into running per-lane argmin:
        // C layout: VGPR r -> row (rowBase + half*8 + r), column nt+s*16+l16.
#pragma unroll
        for (int s = 0; s < 4; ++s) {
            const int   col = nt + s * 16 + l16;
            const float cn  = cnorm[col];
#pragma unroll
            for (int r = 0; r < 8; ++r) {
                float d = fmaf(-2.0f, acc[s][r], cn);
                if (d < minv[r]) { minv[r] = d; mini[r] = col; }
            }
        }
    }

    // cross-lane argmin within each 16-lane column group (xor masks < 16 stay
    // inside the half-wave; halves hold disjoint rows).
#pragma unroll
    for (int m = 1; m < 16; m <<= 1) {
#pragma unroll
        for (int r = 0; r < 8; ++r) {
            float ov = __shfl_xor(minv[r], m, 32);
            int   oi = __shfl_xor(mini[r], m, 32);
            if (ov < minv[r] || (ov == minv[r] && oi < mini[r])) {
                minv[r] = ov; mini[r] = oi;
            }
        }
    }

    if (l16 == 0) {
#pragma unroll
        for (int r = 0; r < 8; ++r)
            out[(size_t)rowBase + half * 8 + r] = (long long)mini[r];
    }
}

// ---------------------------------------------------------------------------
extern "C" void kernel_launch(void* const* d_in, const int* in_sizes, int n_in,
                              void* d_out, int out_size, void* d_ws, size_t ws_size,
                              hipStream_t stream) {
    const float* x = (const float*)d_in[0];   // [16,2000,1024] f32
    const float* c = (const float*)d_in[1];   // [2048,1024]   f32
    long long*   out = (long long*)d_out;     // [16,2000] int64 ids

    char* ws = (char*)d_ws;                   // needs ~139.5 MB
    __bf16* xhi = (__bf16*)(ws + XHI_OFS);
    __bf16* xlo = (__bf16*)(ws + XLO_OFS);
    __bf16* chi = (__bf16*)(ws + CHI_OFS);
    __bf16* clo = (__bf16*)(ws + CLO_OFS);
    float*  cnorm = (float*)(ws + CNORM_OFS);

    kq_split_x<<<32000, 256, 0, stream>>>(x, xhi, xlo);
    kq_split_c<<<256, 256, 0, stream>>>(c, chi, clo, cnorm);
    kq_argmin<<<250, 256, 0, stream>>>(xhi, xlo, chi, clo, cnorm, out);
}
